// G1Policy_63952063037704
// MI455X (gfx1250) — matrix operand order repaired and checked by reading, not compiled, not run on hardware
//
#include <hip/hip_runtime.h>

typedef _Float16 v16h __attribute__((ext_vector_type(16)));
typedef _Float16 v8h  __attribute__((ext_vector_type(8)));
typedef float    v8f  __attribute__((ext_vector_type(8)));

#define SEQ   64
#define BATCH 8192
#define OBS   47
#define HID   64
#define MLPH  32
#define ACT   12
#define PAIRS 2     // row-groups per block (2 waves each)
#define NW    (PAIRS * 2)
#define ROWS  16    // batch rows per group (one WMMA M-tile)

__device__ __forceinline__ float fast_sigmoid(float x) {
  return 1.0f / (1.0f + __expf(-x));
}
__device__ __forceinline__ float fast_tanh(float x) {
  float e = __expf(2.0f * x);
  return (e - 1.0f) / (e + 1.0f);
}

// A-fragment (16x32 f16): lanes 0-15 -> row=lane, K = {kc*32+0..7, kc*32+16..23}
//                         lanes 16-31 -> row=lane-16, K = {kc*32+8..15, kc*32+24..31}
__device__ __forceinline__ v16h ldsA(const _Float16* base, int strideH, int kc, int lane) {
  int r    = lane & 15;
  int koff = kc * 32 + ((lane & 16) ? 8 : 0);
  const _Float16* p = base + r * strideH + koff;
  v8h lo = *(const v8h*)(p);
  v8h hi = *(const v8h*)(p + 16);
  return __builtin_shufflevector(lo, hi, 0,1,2,3,4,5,6,7,8,9,10,11,12,13,14,15);
}

// B-fragment (32x16 f16), weights stored [N][K] row-major in LDS:
// lanes 0-15 -> col=lane, K={kc*32+0..7, kc*32+16..23} ; lanes 16-31 -> col=lane-16, K={kc*32+8..15, kc*32+24..31}
__device__ __forceinline__ v16h ldsB(const _Float16* base, int strideH, int kc, int lane) {
  int n    = lane & 15;
  int koff = kc * 32 + ((lane & 16) ? 8 : 0);
  const _Float16* p = base + n * strideH + koff;
  v8h lo = *(const v8h*)(p);
  v8h hi = *(const v8h*)(p + 16);
  return __builtin_shufflevector(lo, hi, 0,1,2,3,4,5,6,7,8,9,10,11,12,13,14,15);
}

__device__ __forceinline__ v8f wmma_f16(v16h a, v16h b, v8f c) {
  return __builtin_amdgcn_wmma_f32_16x16x32_f16(false, a, false, b, (short)0, c, false, false);
}

__device__ __forceinline__ void lds_drain() {
  asm volatile("s_wait_dscnt 0x0" ::: "memory");
}

__global__ __launch_bounds__(NW * 32, 1)
void lstm_policy_kernel(const float* __restrict__ obs, const float* __restrict__ h0,
                        const float* __restrict__ c0,  const float* __restrict__ Wih,
                        const float* __restrict__ Whh, const float* __restrict__ bih,
                        const float* __restrict__ bhh, const float* __restrict__ W1,
                        const float* __restrict__ b1,  const float* __restrict__ W2,
                        const float* __restrict__ b2,  const int* __restrict__ masks,
                        float* __restrict__ out) {
  // Shared weights (f16, K padded) + per-group tiles.  ~80 KB (<320 KB/WGP).
  __shared__ __align__(16) _Float16 sWih[256 * 64];       // [4H][K=64], K>=47 zero
  __shared__ __align__(16) _Float16 sWhh[256 * 64];       // [4H][64]
  __shared__ __align__(16) _Float16 sW1[32 * 64];         // [32][64]
  __shared__ __align__(16) _Float16 sW2[16 * 32];         // [16][32], rows>=12 zero
  __shared__ float sB[256];
  __shared__ float sB1[32];
  __shared__ float sB2[16];
  __shared__ __align__(16) _Float16 sX[PAIRS][16 * 64];   // obs tile, K>=47 zero
  __shared__ __align__(16) _Float16 sH[PAIRS][16 * 64];   // hidden state tile
  __shared__ __align__(16) _Float16 sA1[PAIRS][16 * 32];  // elu activations
  __shared__ int sM[PAIRS][16];

  const int tid   = threadIdx.x;
  const int bs    = blockDim.x;
  const int lane  = tid & 31;
  // wave-uniform ids in SGPRs -> scalar branches + scalar LDS bases
  const int wuni  = __builtin_amdgcn_readfirstlane(tid) >> 5;
  const int pair  = wuni >> 1;
  const int half  = wuni & 1;               // hidden split: this wave owns j = 2*half, 2*half+1
  const int col   = lane & 15;
  const int hioff = (lane & 16) ? 8 : 0;    // D-layout: VGPR v holds row v + hioff
  const int row0  = (blockIdx.x * PAIRS + pair) * ROWS;
  const int j0    = half * 2;

  // ---- cooperative weight staging (f32 -> f16) ----
  for (int i = tid; i < 256 * 64; i += bs) {
    int n = i >> 6, k = i & 63;
    sWih[i] = (k < OBS) ? (_Float16)Wih[n * OBS + k] : (_Float16)0.f;
    sWhh[i] = (_Float16)Whh[i];
  }
  for (int i = tid; i < 32 * 64; i += bs) sW1[i] = (_Float16)W1[i];
  for (int i = tid; i < 16 * 32; i += bs) {
    int n = i >> 5, k = i & 31;
    sW2[i] = (n < ACT) ? (_Float16)W2[n * 32 + k] : (_Float16)0.f;
  }
  for (int i = tid; i < 256; i += bs) sB[i] = bih[i] + bhh[i];
  if (tid < 32) sB1[tid] = b1[tid];
  if (tid < 16) sB2[tid] = (tid < ACT) ? b2[tid] : 0.f;

  // per-group: zero obs pad columns, stage h0 into f16 LDS tile (pair cooperative)
  for (int i = half * 32 + lane; i < 16 * 64; i += 64) {
    sX[pair][i] = (_Float16)0.f;
    sH[pair][i] = (_Float16)h0[(size_t)row0 * HID + i];
  }

  // cell state (this wave's 2 hidden tiles) in WMMA C/D layout
  v8f c[2];
#pragma unroll
  for (int jj = 0; jj < 2; ++jj)
#pragma unroll
    for (int v = 0; v < 8; ++v)
      c[jj][v] = c0[(size_t)(row0 + v + hioff) * HID + (j0 + jj) * 16 + col];

  lds_drain();
  __syncthreads();

  const size_t hOutBase = (size_t)SEQ * BATCH * ACT;
  const size_t cOutBase = hOutBase + (size_t)BATCH * HID;

#pragma unroll 1
  for (int t = 0; t < SEQ; ++t) {
    // ---- stage x_t / masks_t (readers of previous contents all drained pre-B2(t-1)) ----
    if (half == 0 && lane < 16) sM[pair][lane] = masks[(size_t)t * BATCH + row0 + lane];
    const float* xg = obs + ((size_t)t * BATCH + row0) * OBS;
    for (int i = half * 32 + lane; i < ROWS * OBS; i += 64) {
      int r = i / OBS, k = i - r * OBS;
      sX[pair][r * 64 + k] = (_Float16)xg[i];
    }
    lds_drain();
    __syncthreads();   // B1: x_t, m_t and h_{t-1} visible to both waves of the pair

    // ---- all shared-tile reads for this iteration happen between B1 and B2 ----
    v16h xa0 = ldsA(sX[pair], 64, 0, lane);
    v16h xa1 = ldsA(sX[pair], 64, 1, lane);
    v16h hb0 = ldsA(sH[pair], 64, 0, lane);           // raw h_{t-1} for MLP(t-1)
    v16h hb1 = ldsA(sH[pair], 64, 1, lane);
    const _Float16 mh = (_Float16)(float)sM[pair][col];
    v16h ha0 = hb0 * mh;                              // masked h for gates(t)
    v16h ha1 = hb1 * mh;
    float mD[8];
#pragma unroll
    for (int v = 0; v < 8; ++v) mD[v] = (float)sM[pair][v + hioff];

    // ---- MLP(t-1) part 1: this wave's z1 tile (interleaves with gate WMMAs) ----
    if (t > 0) {
      v8f z;
      float bv = sB1[half * 16 + col];
#pragma unroll
      for (int v = 0; v < 8; ++v) z[v] = bv;
      const _Float16* w1p = sW1 + (half * 16) * 64;
      z = wmma_f16(hb0, ldsB(w1p, 64, 0, lane), z);
      z = wmma_f16(hb1, ldsB(w1p, 64, 1, lane), z);
#pragma unroll
      for (int v = 0; v < 8; ++v) {
        float x = z[v];
        float a = x > 0.f ? x : (__expf(x) - 1.0f);   // elu
        sA1[pair][(v + hioff) * 32 + half * 16 + col] = (_Float16)a;
      }
    }

    // ---- gates(t) for this wave's 2 hidden tiles ----
    v8f acc[2][4];
#pragma unroll
    for (int jj = 0; jj < 2; ++jj) {
      int j = j0 + jj;
#pragma unroll
      for (int g = 0; g < 4; ++g) {
        float bv = sB[g * 64 + j * 16 + col];
#pragma unroll
        for (int v = 0; v < 8; ++v) acc[jj][g][v] = bv;
      }
#pragma unroll
      for (int g = 0; g < 4; ++g) {
        const _Float16* wih = sWih + (g * 64 + j * 16) * 64;
        const _Float16* whh = sWhh + (g * 64 + j * 16) * 64;
        acc[jj][g] = wmma_f16(xa0, ldsB(wih, 64, 0, lane), acc[jj][g]);
        acc[jj][g] = wmma_f16(xa1, ldsB(wih, 64, 1, lane), acc[jj][g]);
        acc[jj][g] = wmma_f16(ha0, ldsB(whh, 64, 0, lane), acc[jj][g]);
        acc[jj][g] = wmma_f16(ha1, ldsB(whh, 64, 1, lane), acc[jj][g]);
      }
    }

    lds_drain();
    __syncthreads();   // B2: all reads of sX/sH/sM done; sA1 halves complete

    // ---- MLP(t-1) part 2: z2 + means store (uniform scalar branch) ----
    if (t > 0 && half == 0) {
      v16h aa = ldsA(sA1[pair], 32, 0, lane);
      v8f z2;
      float bv = sB2[col];
#pragma unroll
      for (int v = 0; v < 8; ++v) z2[v] = bv;
      z2 = wmma_f16(aa, ldsB(sW2, 32, 0, lane), z2);
      if (col < ACT) {
        float* mout = out + ((size_t)(t - 1) * BATCH + row0) * ACT;
#pragma unroll
        for (int v = 0; v < 8; ++v) mout[(v + hioff) * ACT + col] = z2[v];
      }
    }

    // ---- LSTM elementwise update + h_t stores (post-B2: sH safe to overwrite) ----
#pragma unroll
    for (int jj = 0; jj < 2; ++jj) {
      int j = j0 + jj;
#pragma unroll
      for (int v = 0; v < 8; ++v) {
        float cv = c[jj][v] * mD[v];
        float iv = fast_sigmoid(acc[jj][0][v]);
        float fv = fast_sigmoid(acc[jj][1][v]);
        float gv = fast_tanh(acc[jj][2][v]);
        float ov = fast_sigmoid(acc[jj][3][v]);
        cv = fv * cv + iv * gv;
        c[jj][v] = cv;
        float hv = ov * fast_tanh(cv);
        sH[pair][(v + hioff) * 64 + j * 16 + col] = (_Float16)hv;
        if (t == SEQ - 1) {
          size_t r = (size_t)(row0 + v + hioff);
          out[hOutBase + r * HID + j * 16 + col] = hv;
          out[cOutBase + r * HID + j * 16 + col] = cv;
        }
      }
    }
  }

  // ---- epilogue: MLP for t = SEQ-1 ----
  lds_drain();
  __syncthreads();   // h_{63} visible
  {
    v16h hb0 = ldsA(sH[pair], 64, 0, lane);
    v16h hb1 = ldsA(sH[pair], 64, 1, lane);
    v8f z;
    float bv = sB1[half * 16 + col];
#pragma unroll
    for (int v = 0; v < 8; ++v) z[v] = bv;
    const _Float16* w1p = sW1 + (half * 16) * 64;
    z = wmma_f16(hb0, ldsB(w1p, 64, 0, lane), z);
    z = wmma_f16(hb1, ldsB(w1p, 64, 1, lane), z);
#pragma unroll
    for (int v = 0; v < 8; ++v) {
      float x = z[v];
      float a = x > 0.f ? x : (__expf(x) - 1.0f);
      sA1[pair][(v + hioff) * 32 + half * 16 + col] = (_Float16)a;
    }
    lds_drain();
    __syncthreads();
    if (half == 0) {
      v16h aa = ldsA(sA1[pair], 32, 0, lane);
      v8f z2;
      float b2v = sB2[col];
#pragma unroll
      for (int v = 0; v < 8; ++v) z2[v] = b2v;
      z2 = wmma_f16(aa, ldsB(sW2, 32, 0, lane), z2);
      if (col < ACT) {
        float* mout = out + ((size_t)(SEQ - 1) * BATCH + row0) * ACT;
#pragma unroll
        for (int v = 0; v < 8; ++v) mout[(v + hioff) * ACT + col] = z2[v];
      }
    }
  }
}

extern "C" void kernel_launch(void* const* d_in, const int* in_sizes, int n_in,
                              void* d_out, int out_size, void* d_ws, size_t ws_size,
                              hipStream_t stream) {
  (void)in_sizes; (void)n_in; (void)out_size; (void)d_ws; (void)ws_size;
  const float* obs  = (const float*)d_in[0];
  const float* h0   = (const float*)d_in[1];
  const float* c0   = (const float*)d_in[2];
  const float* Wih  = (const float*)d_in[3];
  const float* Whh  = (const float*)d_in[4];
  const float* bih  = (const float*)d_in[5];
  const float* bhh  = (const float*)d_in[6];
  const float* W1   = (const float*)d_in[7];
  const float* b1   = (const float*)d_in[8];
  const float* W2   = (const float*)d_in[9];
  const float* b2   = (const float*)d_in[10];
  const int*   mk   = (const int*)d_in[11];
  float* out = (float*)d_out;

  dim3 grid(BATCH / (PAIRS * ROWS));  // 256 blocks
  dim3 block(NW * 32);                // 4 waves (2 pairs)
  lstm_policy_kernel<<<grid, block, 0, stream>>>(obs, h0, c0, Wih, Whh, bih, bhh,
                                                 W1, b1, W2, b2, mk, out);
}
